// MultiheadDiffAttn_83193516523983
// MI455X (gfx1250) — matrix operand order repaired and checked by reading, not compiled, not run on hardware
//
#include <hip/hip_runtime.h>

typedef __bf16 bf16;
typedef __attribute__((ext_vector_type(16))) __bf16 v16bf;
typedef __attribute__((ext_vector_type(8)))  __bf16 v8bf;
typedef __attribute__((ext_vector_type(4)))  __bf16 v4bf;
typedef __attribute__((ext_vector_type(8)))  float  v8f;
typedef __attribute__((ext_vector_type(4)))  float  v4f;

#define S_LEN 2048
#define D_EMB 1024
#define NH    16
#define NHC   32      // 2*NUM_HEADS head-components
#define HD    32      // head dim per component (== WMMA K for bf16!)
#define VD    64      // value dim per real head
#define LAMBDA_INIT 0.7836057665316244f
#define EPS_RMS 1e-5f
#define THETA_ROPE 1e-4f

union AB16 { v16bf v; v8bf h[2]; v4bf q[4]; };

// ---------------------------------------------------------------- cvt f32->bf16
__global__ void f32_to_bf16_kernel(const float* __restrict__ in, bf16* __restrict__ out, int n) {
    int i = blockIdx.x * blockDim.x + threadIdx.x;
    if (i < n) out[i] = (bf16)in[i];
}

// ------------------------------------------------- transpose + cvt: out[n][k] = in[k][n]
__global__ void transpose_cvt_kernel(const float* __restrict__ in, bf16* __restrict__ out,
                                     int R /*rows of in*/, int C /*cols of in*/) {
    __shared__ float tile[32][33];
    int c = blockIdx.x * 32 + threadIdx.x;
    int r = blockIdx.y * 32 + threadIdx.y;
    #pragma unroll
    for (int i = 0; i < 32; i += 8)
        tile[threadIdx.y + i][threadIdx.x] = in[(size_t)(r + i) * C + c];
    __syncthreads();
    int orow = blockIdx.x * 32 + threadIdx.y;   // output row = original col
    int oc   = blockIdx.y * 32 + threadIdx.x;   // output col = original row
    #pragma unroll
    for (int i = 0; i < 32; i += 8)
        out[(size_t)(orow + i) * R + oc] = (bf16)tile[threadIdx.x][threadIdx.y + i];
}

// ---------------------------------------------------------------- WMMA GEMM
// C[M,N] (f32) = A[M,K] (bf16, row-major) * Bt[N,K]^T (bf16, row-major = B transposed)
// block = 256 threads = 8 waves; each wave computes a 16x64 strip (4 accumulators,
// A fragment reused 4x per k-step).  grid = (N/64, M/128)
__global__ void gemm_bf16_wmma(const bf16* __restrict__ A, const bf16* __restrict__ Bt,
                               float* __restrict__ C, int M, int N, int K) {
    const int lane = threadIdx.x & 31;
    const int wave = threadIdx.x >> 5;
    const int m0 = (blockIdx.y * 8 + wave) * 16;
    const int n0 = blockIdx.x * 64;
    // A fragment: lane L -> row m0+(L&15); halves hold K blocks offset by (L>=16 ? 8 : 0)
    const bf16* pa  = A + (size_t)(m0 + (lane & 15)) * K + ((lane >> 4) * 8);
    // B fragment: lane L -> col (L&15) of tile; halves hold K blocks offset by (L>=16 ? 8 : 0)
    const bf16* pb0 = Bt + (size_t)(n0 + (lane & 15)) * K + ((lane >> 4) * 8);
    v8f acc[4] = {};
    for (int k0 = 0; k0 < K; k0 += 32) {
        __builtin_prefetch(pa + k0 + 512, 0, 1);     // global_prefetch_b8, next K blocks
        __builtin_prefetch(pb0 + k0 + 512, 0, 1);
        AB16 a;
        a.h[0] = *(const v8bf*)(pa + k0);
        a.h[1] = *(const v8bf*)(pa + k0 + 16);
        #pragma unroll
        for (int t = 0; t < 4; ++t) {
            AB16 b;
            const bf16* pb = pb0 + (size_t)t * 16 * K + k0;
            b.h[0] = *(const v8bf*)(pb);
            b.h[1] = *(const v8bf*)(pb + 16);
            acc[t] = __builtin_amdgcn_wmma_f32_16x16x32_bf16(false, a.v, false, b.v,
                                                             (short)0, acc[t], false, false);
        }
    }
    const int rbase = m0 + ((lane >> 4) << 3);
    const int colb  = lane & 15;
    #pragma unroll
    for (int t = 0; t < 4; ++t) {
        const int col = n0 + t * 16 + colb;
        #pragma unroll
        for (int r = 0; r < 8; ++r)
            C[(size_t)(rbase + r) * N + col] = acc[t][r];
    }
}

// ------------------------------------------- RoPE (interleaved pairs) + scale + repack
// in:  f32 (s, 1024) = (s, hc=32, hd=32) ; out: bf16 [hc][s][32]
__global__ void rope_cvt_kernel(const float* __restrict__ in, bf16* __restrict__ out, float scale) {
    int i = blockIdx.x * blockDim.x + threadIdx.x;      // (s, hc, pair)
    if (i >= S_LEN * NHC * (HD / 2)) return;
    int p  = i & 15;
    int hc = (i >> 4) & 31;
    int s  = i >> 9;
    const float* src = in + (size_t)s * D_EMB + hc * HD + 2 * p;
    float q0 = src[0], q1 = src[1];
    float freq = powf(THETA_ROPE, -(float)p / 16.0f);   // 1/theta^(2p/d), d=32
    float ang = (float)s * freq;
    float cs = cosf(ang), sn = sinf(ang);
    bf16* o = out + (size_t)hc * (S_LEN * HD) + (size_t)s * HD + 2 * p;
    o[0] = (bf16)((q0 * cs - q1 * sn) * scale);
    o[1] = (bf16)((q0 * sn + q1 * cs) * scale);
}

// ---------------------------------------------------- V transpose: Vt[h][d][s] = V[s][h*64+d]
__global__ void v_transpose_kernel(const float* __restrict__ Vf, bf16* __restrict__ Vt) {
    int i = blockIdx.x * blockDim.x + threadIdx.x;      // h*64*2048 + d*2048 + s
    if (i >= NH * VD * S_LEN) return;
    int s = i & (S_LEN - 1);
    int d = (i >> 11) & 63;
    int h = i >> 17;
    Vt[i] = (bf16)Vf[(size_t)s * D_EMB + h * VD + d];
}

// ---------------------------------------------------------------- lambda scalar
__global__ void lambda_kernel(const float* lq1, const float* lk1,
                              const float* lq2, const float* lk2, float* lam) {
    int lane = threadIdx.x;
    float a = lq1[lane] * lk1[lane];
    float b = lq2[lane] * lk2[lane];
    #pragma unroll
    for (int m = 16; m >= 1; m >>= 1) {
        a += __shfl_xor(a, m, 32);
        b += __shfl_xor(b, m, 32);
    }
    if (lane == 0) *lam = expf(a) - expf(b) + LAMBDA_INIT;
}

// ---------------------------------------------------------------- attention core
// One block = 16 queries of one head-component. Scores strip (16 x 2048 f32) lives in LDS.
// grid = (32 head-comps, 128 q-blocks), block = 256 (8 waves), dyn LDS ~136KB (WGP has 320KB).
__global__ void attn_kernel(const bf16* __restrict__ Qb, const bf16* __restrict__ Kb,
                            const bf16* __restrict__ Vt, float* __restrict__ O) {
    extern __shared__ float smem[];
    float* Ssc    = smem;                  // [16][2048] scores -> exp(S-max) in place
    float* Oscr   = Ssc + 16 * S_LEN;      // [8][256] per-wave partial O tiles
    float* rowsum = Oscr + 8 * 256;        // [16]
    const int hc   = blockIdx.x;
    const int q0   = blockIdx.y * 16;
    const int lane = threadIdx.x & 31;
    const int wave = threadIdx.x >> 5;
    const bf16* Qh = Qb + (size_t)hc * (S_LEN * HD);
    const bf16* Kh = Kb + (size_t)hc * (S_LEN * HD);
    const bf16* Vh = Vt + (size_t)(hc >> 1) * (VD * S_LEN);

    // Q tile A-fragment (K = head_dim = 32 exactly -> one fragment, loaded once)
    AB16 aq;
    {
        const bf16* pa = Qh + (size_t)(q0 + (lane & 15)) * HD + ((lane >> 4) * 8);
        aq.h[0] = *(const v8bf*)(pa);
        aq.h[1] = *(const v8bf*)(pa + 16);
    }

    // Phase 1: scores S[16][2048], one WMMA per 16-key tile, waves stripe over tiles
    for (int kt = wave; kt < S_LEN / 16; kt += 8) {
        const int n0 = kt * 16;
        AB16 bk;
        const bf16* pb = Kh + (size_t)(n0 + (lane & 15)) * HD + ((lane >> 4) * 8);
        __builtin_prefetch(pb + 128 * HD, 0, 1);    // next tile for this wave
        bk.h[0] = *(const v8bf*)(pb);
        bk.h[1] = *(const v8bf*)(pb + 16);
        v8f c = {};
        c = __builtin_amdgcn_wmma_f32_16x16x32_bf16(false, aq.v, false, bk.v,
                                                    (short)0, c, false, false);
        const int rbase = (lane >> 4) << 3;
        const int col   = n0 + (lane & 15);
        #pragma unroll
        for (int r = 0; r < 8; ++r) Ssc[(rbase + r) * S_LEN + col] = c[r];
    }
    __syncthreads();

    // Phase 2: softmax (un-normalized): store exp(S-max) in place, rowsum to LDS.
    #pragma unroll
    for (int rr = 0; rr < 2; ++rr) {
        const int row = wave * 2 + rr;
        float* Srow = Ssc + row * S_LEN;
        float mx = -3.4e38f;
        for (int k = lane; k < S_LEN; k += 32) mx = fmaxf(mx, Srow[k]);
        #pragma unroll
        for (int m = 16; m >= 1; m >>= 1) mx = fmaxf(mx, __shfl_xor(mx, m, 32));
        float sum = 0.f;
        for (int k = lane; k < S_LEN; k += 32) {
            float e = expf(Srow[k] - mx);
            Srow[k] = e;
            sum += e;
        }
        #pragma unroll
        for (int m = 16; m >= 1; m >>= 1) sum += __shfl_xor(sum, m, 32);
        if (lane == 0) rowsum[row] = sum;
    }
    __syncthreads();

    // Phase 3: O = P @ V.  wave -> dim-tile (wave&3, 16 dims), key-half (wave>>2).
    const int n0   = (wave & 3) * 16;
    const int kbeg = (wave >> 2) * (S_LEN / 2);
    v8f acc = {};
    for (int k0 = kbeg; k0 < kbeg + S_LEN / 2; k0 += 32) {
        AB16 ap;
        const float* ps = Ssc + (lane & 15) * S_LEN + k0 + ((lane >> 4) * 8);
        // vectorized LDS reads + packed f32->bf16 conversion
        ap.q[0] = __builtin_convertvector(*(const v4f*)(ps),      v4bf);
        ap.q[1] = __builtin_convertvector(*(const v4f*)(ps + 4),  v4bf);
        ap.q[2] = __builtin_convertvector(*(const v4f*)(ps + 16), v4bf);
        ap.q[3] = __builtin_convertvector(*(const v4f*)(ps + 20), v4bf);
        AB16 bv;
        const bf16* pb = Vh + (size_t)(n0 + (lane & 15)) * S_LEN + k0 + ((lane >> 4) * 8);
        bv.h[0] = *(const v8bf*)(pb);
        bv.h[1] = *(const v8bf*)(pb + 16);
        acc = __builtin_amdgcn_wmma_f32_16x16x32_bf16(false, ap.v, false, bv.v,
                                                      (short)0, acc, false, false);
    }
    float* my = Oscr + wave * 256;
    #pragma unroll
    for (int r = 0; r < 8; ++r) my[r * 32 + lane] = acc[r];
    __syncthreads();

    // Reduce key-halves, fold in 1/rowsum, write O[hc][q][64]
    if (wave < 4) {
        const int rbase = (lane >> 4) << 3;
        const int col   = wave * 16 + (lane & 15);
        #pragma unroll
        for (int r = 0; r < 8; ++r) {
            const int row = rbase + r;
            float v = Oscr[wave * 256 + r * 32 + lane] + Oscr[(wave + 4) * 256 + r * 32 + lane];
            v *= (1.0f / rowsum[row]);
            O[((size_t)hc * S_LEN + q0 + row) * VD + col] = v;
        }
    }
}

// ------------------------------- diff combine + RMSNorm: one wave per (h, s) row of 64
__global__ void combine_kernel(const float* __restrict__ O, const float* __restrict__ lamp,
                               const float* __restrict__ g, bf16* __restrict__ attnb) {
    const int lane = threadIdx.x & 31;
    const int wave = threadIdx.x >> 5;
    const int idx  = blockIdx.x * 8 + wave;     // h*2048 + s
    const int h = idx >> 11;
    const int s = idx & (S_LEN - 1);
    const float lam = *lamp;
    const float* o1 = O + ((size_t)(2 * h)     * S_LEN + s) * VD;
    const float* o2 = O + ((size_t)(2 * h + 1) * S_LEN + s) * VD;
    float v0 = o1[lane]      - lam * o2[lane];
    float v1 = o1[lane + 32] - lam * o2[lane + 32];
    float ss = v0 * v0 + v1 * v1;
    #pragma unroll
    for (int m = 16; m >= 1; m >>= 1) ss += __shfl_xor(ss, m, 32);
    float rms = rsqrtf(ss * (1.0f / VD) + EPS_RMS);
    const float os = 1.0f - LAMBDA_INIT;
    bf16* out = attnb + (size_t)s * D_EMB + h * VD;
    out[lane]      = (bf16)(v0 * rms * g[lane]      * os);
    out[lane + 32] = (bf16)(v1 * rms * g[lane + 32] * os);
}

// ================================================================ host launcher
extern "C" void kernel_launch(void* const* d_in, const int* in_sizes, int n_in,
                              void* d_out, int out_size, void* d_ws, size_t ws_size,
                              hipStream_t stream) {
    const float* x   = (const float*)d_in[0];
    const float* Wq  = (const float*)d_in[1];
    const float* Wk  = (const float*)d_in[2];
    const float* Wv  = (const float*)d_in[3];
    const float* Wo  = (const float*)d_in[4];
    const float* lq1 = (const float*)d_in[5];
    const float* lk1 = (const float*)d_in[6];
    const float* lq2 = (const float*)d_in[7];
    const float* lk2 = (const float*)d_in[8];
    const float* sg  = (const float*)d_in[9];

    char* ws = (char*)d_ws;
    auto alloc = [&](size_t bytes) -> char* {
        char* p = ws;
        ws += (bytes + 255) & ~(size_t)255;
        return p;
    };
    bf16*  xb    = (bf16*)alloc((size_t)S_LEN * D_EMB * 2);
    bf16*  Wqt   = (bf16*)alloc((size_t)D_EMB * D_EMB * 2);
    bf16*  Wkt   = (bf16*)alloc((size_t)D_EMB * D_EMB * 2);
    bf16*  Wvt   = (bf16*)alloc((size_t)D_EMB * D_EMB * 2);
    bf16*  Wot   = (bf16*)alloc((size_t)D_EMB * D_EMB * 2);
    float* Qf    = (float*)alloc((size_t)S_LEN * D_EMB * 4);
    float* Kf    = (float*)alloc((size_t)S_LEN * D_EMB * 4);
    float* Vf    = (float*)alloc((size_t)S_LEN * D_EMB * 4);
    bf16*  Qb    = (bf16*)alloc((size_t)NHC * S_LEN * HD * 2);
    bf16*  Kb    = (bf16*)alloc((size_t)NHC * S_LEN * HD * 2);
    bf16*  Vt    = (bf16*)alloc((size_t)NH * VD * S_LEN * 2);
    float* O     = (float*)alloc((size_t)NHC * S_LEN * VD * 4);
    bf16*  attnb = (bf16*)alloc((size_t)S_LEN * D_EMB * 2);
    float* lam   = (float*)alloc(256);

    // 1) convert x, transpose-convert weights to bf16
    f32_to_bf16_kernel<<<(S_LEN * D_EMB) / 256, 256, 0, stream>>>(x, xb, S_LEN * D_EMB);
    dim3 tb(32, 8), tg(D_EMB / 32, D_EMB / 32);
    transpose_cvt_kernel<<<tg, tb, 0, stream>>>(Wq, Wqt, D_EMB, D_EMB);
    transpose_cvt_kernel<<<tg, tb, 0, stream>>>(Wk, Wkt, D_EMB, D_EMB);
    transpose_cvt_kernel<<<tg, tb, 0, stream>>>(Wv, Wvt, D_EMB, D_EMB);
    transpose_cvt_kernel<<<tg, tb, 0, stream>>>(Wo, Wot, D_EMB, D_EMB);

    // 2) projection GEMMs (WMMA, 16x64 per wave)
    dim3 ggrid(D_EMB / 64, S_LEN / 128);
    gemm_bf16_wmma<<<ggrid, 256, 0, stream>>>(xb, Wqt, Qf, S_LEN, D_EMB, D_EMB);
    gemm_bf16_wmma<<<ggrid, 256, 0, stream>>>(xb, Wkt, Kf, S_LEN, D_EMB, D_EMB);
    gemm_bf16_wmma<<<ggrid, 256, 0, stream>>>(xb, Wvt, Vf, S_LEN, D_EMB, D_EMB);

    // 3) RoPE + repack head-major (Q scaled by hd^-1/2), V transpose
    int nrope = S_LEN * NHC * (HD / 2);
    rope_cvt_kernel<<<(nrope + 255) / 256, 256, 0, stream>>>(Qf, Qb, 0.17677669529663689f);
    rope_cvt_kernel<<<(nrope + 255) / 256, 256, 0, stream>>>(Kf, Kb, 1.0f);
    v_transpose_kernel<<<(NH * VD * S_LEN) / 256, 256, 0, stream>>>(Vf, Vt);
    lambda_kernel<<<1, 32, 0, stream>>>(lq1, lk1, lq2, lk2, lam);

    // 4) attention (LDS-resident score strips, WMMA both matmuls)
    size_t smem = (size_t)(16 * S_LEN + 8 * 256 + 16) * sizeof(float);
    attn_kernel<<<dim3(NHC, S_LEN / 16), 256, smem, stream>>>(Qb, Kb, Vt, O);

    // 5) differential combine + RMSNorm, then output GEMM (WMMA)
    combine_kernel<<<(NH * S_LEN) / 8, 256, 0, stream>>>(O, lam, sg, attnb);
    gemm_bf16_wmma<<<dim3(D_EMB / 64, S_LEN / 128), 256, 0, stream>>>(attnb, Wot,
                                                                      (float*)d_out,
                                                                      S_LEN, D_EMB, D_EMB);
}
